// AutomatonNetwork_55104430408157
// MI455X (gfx1250) — matrix operand order repaired and checked by reading, not compiled, not run on hardware
//
#include <hip/hip_runtime.h>
#include <hip/hip_bf16.h>

// ---------------------------------------------------------------------------
// AutomatonNetwork forward pass for MI455X (gfx1250), chunked matrix-scan form.
//
//   v_{t+1} = v_t @ M[seq[t]],  prob *= dot(v_t, p[seq[t]])
//
// Phase 1 (hot): per-chunk matrix products, register-resident transposed form:
//     R = P^T,  R_t = M_t^T @ R_{t-1}
//   Wave w owns columns [16w,16w+16) of R in 4 WMMA accumulator tiles.
//   A operand (M^T) is pre-swizzled per symbol into exact A-fragment order ->
//   contiguous 32B loads. D->B conversion is a half-wave __shfl_xor + cndmask
//   + cvt, all in registers. No LDS, no barriers in the hot loop.
// Phase 2: hierarchical fp32 vector scan -> exact chunk-entry vectors u_c.
// Phase 3: per-chunk fp32 replay accumulating sum(log(dot(v,p))).
// Phase 4: deterministic fixed-order log reduction + final dot.
// ---------------------------------------------------------------------------

typedef __attribute__((ext_vector_type(16))) _Float16 v16h;
typedef __attribute__((ext_vector_type(8)))  float    v8f;

#define T_LEN     524288
#define SDIM      64
#define MAT_ELEMS (SDIM * SDIM)      // 4096
#define CHUNKS    2048
#define CLEN      (T_LEN / CHUNKS)   // 256
#define NSUP      32
#define SUPG      (CHUNKS / NSUP)    // 64

// ---- workspace layout (bytes) ---------------------------------------------
#define OFF_TA     ((size_t)0)                                   // 128*4096 f16 (A-swizzled M^T) = 1 MB
#define OFF_PRODS  (OFF_TA     + (size_t)128 * MAT_ELEMS * 2)    // CHUNKS*4096 f32 = 32 MB
#define OFF_SUPERS (OFF_PRODS  + (size_t)CHUNKS * MAT_ELEMS * 4) // NSUP*4096 f32
#define OFF_USUP   (OFF_SUPERS + (size_t)NSUP * MAT_ELEMS * 4)   // NSUP*64 f32
#define OFF_UVEC   (OFF_USUP   + (size_t)NSUP * SDIM * 4)        // CHUNKS*64 f32
#define OFF_LOGS   (OFF_UVEC   + (size_t)CHUNKS * SDIM * 4)      // CHUNKS f32
#define OFF_VFIN   (OFF_LOGS   + (size_t)CHUNKS * 4)             // 64 f32

// ---------------------------------------------------------------------------
// WMMA fragment layouts used here (wave32; contiguous k order, not the ISA 7.12.2 table order):
// A 16x32 f16 : lane l: row m=l%16; halves h=0..15 -> K=base0+h,
//               base0 = (l<16 ? 0 : 16).
// B 32x16 f16 : lane l: col n=l%16; half h -> K = (l<16?0:16)+h.
// C/D 16x16 f32: vgpr j at lane l: row = j + (l>=16?8:0), col = l%16.
// ---------------------------------------------------------------------------

// ---------------------------------------------------------------------------
// Kernel 0: build A-fragment-swizzled f16 image of M_s^T for every symbol s.
// transA[s][f][l][h], f = it*2+kb:  value = M_s^T[16*it + (l%16)][kb*32 + K(h,l)]
//                                        = trans[s][K(h,l)_global][16*it + (l%16)]
// Each lane's 16 halves for a fragment are contiguous (32 B) -> b128 loads.
// ---------------------------------------------------------------------------
__global__ void k_convertA(const float* __restrict__ trans, _Float16* __restrict__ transA) {
  int idx = blockIdx.x * blockDim.x + threadIdx.x;
  if (idx >= 128 * MAT_ELEMS) return;
  const int h = idx & 15;
  const int l = (idx >> 4) & 31;
  const int f = (idx >> 9) & 7;
  const int s = idx >> 12;
  const int it = f >> 1, kb = f & 1;
  const int i = it * 16 + (l & 15);                 // row of M^T  (= col of M)
  const int base0 = (l < 16) ? 0 : 16;
  const int K = kb * 32 + base0 + h;  // col of M^T (= row of M)
  transA[idx] = (_Float16)trans[(size_t)s * MAT_ELEMS + K * SDIM + i];
}

// ---------------------------------------------------------------------------
// Kernel 1 (hot): per-chunk product, register-resident.
//   R tiles (it=0..3): R[it].v[j] = P^T[16*it + j + (l>=16?8:0)][16*w + l%16]
// ---------------------------------------------------------------------------
__global__ __launch_bounds__(128) void k_chunk_prod(const float* __restrict__ trans,
                                                    const _Float16* __restrict__ transA,
                                                    const int* __restrict__ seq,
                                                    float* __restrict__ prods) {
  const int c = blockIdx.x, tid = threadIdx.x;
  const int lane = tid & 31, wv = tid >> 5;
  const int base = c * CLEN;
  const bool lo = lane < 16;
  const int nlo = lane & 15;

  // ---- init: R = M_{seq[base]}^T restricted to our column block -----------
  v8f R[4];
  {
    const float* M0 = trans + (size_t)seq[base] * MAT_ELEMS + (16 * wv + nlo) * SDIM;
#pragma unroll
    for (int it = 0; it < 4; ++it) {
      const float* src = M0 + 16 * it + (lo ? 0 : 8);
#pragma unroll
      for (int j = 0; j < 8; ++j) R[it][j] = src[j];   // 8 contiguous floats
    }
  }

  const v8f vzero = {0.f, 0.f, 0.f, 0.f, 0.f, 0.f, 0.f, 0.f};

  for (int t = 1; t < CLEN; ++t) {
    // ---- A fragments of M_t^T: 8 x 32B contiguous vector loads ------------
    const _Float16* As = transA + (size_t)seq[base + t] * MAT_ELEMS;
    v16h afr[8];
#pragma unroll
    for (int f = 0; f < 8; ++f)
      afr[f] = *(const v16h*)(As + ((f * 32 + lane) << 4));

    if (t + 1 < CLEN) {  // prefetch next symbol's 8 KB A image (global_prefetch_b8)
      const char* nxt = (const char*)(transA + (size_t)seq[base + t + 1] * MAT_ELEMS);
#pragma unroll
      for (int k = 0; k < 4; ++k) __builtin_prefetch(nxt + lane * 256 + k * 64, 0, 1);
    }

    // ---- build B fragments (R_{t-1}, f16) from f32 register tiles ---------
    // b_kb[h]   = R[32*kb + (l<16?0:16) + h][own col],  h = 0..15
    // rows 0..7 of a 16-row group live in lanes 0..15, rows 8..15 in 16..31:
    // one half-wave swap per element pair.
    v16h bfr[2];
#pragma unroll
    for (int kb = 0; kb < 2; ++kb) {
      const v8f& R0 = R[2 * kb];
      const v8f& R1 = R[2 * kb + 1];
#pragma unroll
      for (int h = 0; h < 8; ++h) {
        float z = lo ? R1[h] : R0[h];          // value my partner lane needs
        float s = __shfl_xor(z, 16, 32);       // cross half-wave exchange
        float bl = lo ? R0[h] : s;             // b[h]   : rows 0..7 of group
        float bh = lo ? s : R1[h];             // b[h+8] : rows 8..15 of group
        bfr[kb][h]     = (_Float16)bl;
        bfr[kb][h + 8] = (_Float16)bh;
      }
    }

    // ---- R_t = M_t^T @ R_{t-1} : 8 WMMAs --------------------------------
    v8f acc[4];
#pragma unroll
    for (int it = 0; it < 4; ++it) acc[it] = vzero;
#pragma unroll
    for (int kb = 0; kb < 2; ++kb)
#pragma unroll
      for (int it = 0; it < 4; ++it)
        acc[it] = __builtin_amdgcn_wmma_f32_16x16x32_f16(
            false, afr[it * 2 + kb], false, bfr[kb], (short)0, acc[it], false, false);
#pragma unroll
    for (int it = 0; it < 4; ++it) R[it] = acc[it];
  }

  // ---- store chunk product as plain row-major P (un-transpose is free:
  //      per lane the 8 j-values are contiguous in P's row) -----------------
  float* dst = prods + (size_t)c * MAT_ELEMS + (16 * wv + nlo) * SDIM;
#pragma unroll
  for (int it = 0; it < 4; ++it) {
    float* d = dst + 16 * it + (lo ? 0 : 8);
#pragma unroll
    for (int j = 0; j < 8; ++j) d[j] = R[it][j];
  }
}

// ---------------------------------------------------------------------------
// LDS-staged helpers for the tiny super-product kernel (round-0 style).
// ---------------------------------------------------------------------------
__device__ __forceinline__ v16h lds_load_A(const _Float16* P, int rowblk, int kb, int lane) {
  const int m = rowblk * 16 + (lane & 15);
  const int base0 = (lane < 16) ? 0 : 16;
  const _Float16* row = P + m * SDIM + kb * 32;
  v16h a;
#pragma unroll
  for (int h = 0; h < 8; ++h) a[h] = row[base0 + h];
#pragma unroll
  for (int h = 0; h < 8; ++h) a[h + 8] = row[base0 + 8 + h];
  return a;
}

__device__ __forceinline__ v16h lds_load_B(const _Float16* M, int kb, int nt, int lane) {
  const int n = nt * 16 + (lane & 15);
  const int kbase = kb * 32 + ((lane < 16) ? 0 : 16);
  v16h b;
#pragma unroll
  for (int h = 0; h < 16; ++h) b[h] = M[(kbase + h) * SDIM + n];
  return b;
}

// ---------------------------------------------------------------------------
// Kernel 2a: super products  W_b = P_{bG} @ ... @ P_{bG+G-1}.  (<1% of work)
// ---------------------------------------------------------------------------
__global__ __launch_bounds__(128) void k_super_prod(const float* __restrict__ prods,
                                                    float* __restrict__ supers) {
  __shared__ __align__(16) _Float16 Pb[2][MAT_ELEMS];
  __shared__ __align__(16) _Float16 Mb[MAT_ELEMS];
  const int b = blockIdx.x, tid = threadIdx.x;
  const int lane = tid & 31, wv = tid >> 5;

  const float* first = prods + (size_t)b * SUPG * MAT_ELEMS;
  for (int i = tid; i < MAT_ELEMS; i += 128) Pb[0][i] = (_Float16)first[i];
  __syncthreads();

  const v8f vzero = {0.f, 0.f, 0.f, 0.f, 0.f, 0.f, 0.f, 0.f};
  v8f acc[4] = {vzero, vzero, vzero, vzero};
  int cur = 0;
  for (int j = 1; j < SUPG; ++j) {
    const float* Msrc = prods + (size_t)(b * SUPG + j) * MAT_ELEMS;
    for (int i = tid; i < MAT_ELEMS; i += 128) Mb[i] = (_Float16)Msrc[i];
    __syncthreads();
#pragma unroll
    for (int nt = 0; nt < 4; ++nt) acc[nt] = vzero;
#pragma unroll
    for (int kb = 0; kb < 2; ++kb) {
      v16h a = lds_load_A(&Pb[cur][0], wv, kb, lane);
#pragma unroll
      for (int nt = 0; nt < 4; ++nt) {
        v16h bb = lds_load_B(&Mb[0], kb, nt, lane);
        acc[nt] = __builtin_amdgcn_wmma_f32_16x16x32_f16(
            false, a, false, bb, (short)0, acc[nt], false, false);
      }
    }
    // write back f16 for next iteration's A
    {
      const int nlo = lane & 15;
      const int mb = wv * 16 + ((lane < 16) ? 0 : 8);
#pragma unroll
      for (int nt = 0; nt < 4; ++nt)
#pragma unroll
        for (int jj = 0; jj < 8; ++jj)
          Pb[cur ^ 1][(mb + jj) * SDIM + nt * 16 + nlo] = (_Float16)acc[nt][jj];
    }
    cur ^= 1;
    __syncthreads();
  }
  {  // final product from f32 accumulators
    const int nlo = lane & 15;
    const int mb = wv * 16 + ((lane < 16) ? 0 : 8);
    float* dst = supers + (size_t)b * MAT_ELEMS;
#pragma unroll
    for (int nt = 0; nt < 4; ++nt)
#pragma unroll
      for (int jj = 0; jj < 8; ++jj)
        dst[(mb + jj) * SDIM + nt * 16 + nlo] = acc[nt][jj];
  }
}

// ---------------------------------------------------------------------------
// Kernel 2b: sequential fp32 scan over the 32 super products.
// ---------------------------------------------------------------------------
__global__ void k_super_scan(const float* __restrict__ start_vec,
                             const float* __restrict__ supers,
                             float* __restrict__ usuper) {
  __shared__ float sv[SDIM];
  const int tid = threadIdx.x;
  sv[tid] = start_vec[tid];
  __syncthreads();
  for (int b = 0; b < NSUP; ++b) {
    usuper[b * SDIM + tid] = sv[tid];
    const float* W = supers + (size_t)b * MAT_ELEMS;
    float acc = 0.f;
    for (int m = 0; m < SDIM; ++m) acc += sv[m] * W[m * SDIM + tid];
    __syncthreads();
    sv[tid] = acc;
    __syncthreads();
  }
}

// ---------------------------------------------------------------------------
// Kernel 2c: fp32 scan over chunk products within each super-group.
// ---------------------------------------------------------------------------
__global__ void k_uvec_fill(const float* __restrict__ usuper,
                            const float* __restrict__ prods,
                            float* __restrict__ uvec) {
  __shared__ float sv[SDIM];
  const int b = blockIdx.x, tid = threadIdx.x;
  sv[tid] = usuper[b * SDIM + tid];
  __syncthreads();
  for (int j = 0; j < SUPG; ++j) {
    const int c = b * SUPG + j;
    uvec[(size_t)c * SDIM + tid] = sv[tid];
    if (j == SUPG - 1) break;
    const float* P = prods + (size_t)c * MAT_ELEMS;
    float acc = 0.f;
    for (int m = 0; m < SDIM; ++m) acc += sv[m] * P[m * SDIM + tid];
    __syncthreads();
    sv[tid] = acc;
    __syncthreads();
  }
}

// ---------------------------------------------------------------------------
// Kernel 3: per-chunk fp32 replay; dot BEFORE transition, like the reference.
// ---------------------------------------------------------------------------
__global__ void k_emit(const int* __restrict__ seq, const float* __restrict__ trans,
                       const float* __restrict__ probt, const float* __restrict__ uvec,
                       float* __restrict__ logs, float* __restrict__ vfinal) {
  __shared__ float sv[SDIM];
  const int c = blockIdx.x, tid = threadIdx.x;
  const int base = c * CLEN;
  sv[tid] = uvec[(size_t)c * SDIM + tid];
  __syncthreads();

  float sumlog = 0.f;  // meaningful in thread 0 only
  for (int t = 0; t < CLEN; ++t) {
    const int sym = seq[base + t];
    const float* M = trans + (size_t)sym * MAT_ELEMS;
    const float* p = probt + sym * SDIM;

    if (tid < 32) {  // wave 0: dot(v, p) via intra-wave shuffle reduction
      float fp = sv[tid] * p[tid] + sv[tid + 32] * p[tid + 32];
      fp += __shfl_xor(fp, 16, 32);
      fp += __shfl_xor(fp, 8, 32);
      fp += __shfl_xor(fp, 4, 32);
      fp += __shfl_xor(fp, 2, 32);
      fp += __shfl_xor(fp, 1, 32);
      if (tid == 0) sumlog += __logf(fp);
    }
    if (t + 1 < CLEN) {  // prefetch next 16 KB matrix
      const float* Mn = trans + (size_t)seq[base + t + 1] * MAT_ELEMS;
#pragma unroll
      for (int k = 0; k < 4; ++k)
        __builtin_prefetch((const char*)Mn + (tid + k * 64) * 64, 0, 1);
    }
    float acc = 0.f;  // v_new[tid] = sum_m v[m] * M[m][tid]
#pragma unroll 8
    for (int m = 0; m < SDIM; ++m) acc += sv[m] * M[m * SDIM + tid];
    __syncthreads();
    sv[tid] = acc;
    __syncthreads();
  }
  if (tid == 0) logs[c] = sumlog;
  if (c == CHUNKS - 1) vfinal[tid] = sv[tid];  // state after all T steps
}

// ---------------------------------------------------------------------------
// Kernel 4: deterministic fixed-order reduction of chunk logs + final dot.
// ---------------------------------------------------------------------------
__global__ void k_final(const float* __restrict__ logs, const float* __restrict__ vfinal,
                        const float* __restrict__ finals, const float* __restrict__ start_prob,
                        float* __restrict__ out) {
  __shared__ float red[SDIM];
  const int tid = threadIdx.x;
  float s = 0.f;
  for (int i = tid; i < CHUNKS; i += SDIM) s += logs[i];
  red[tid] = s;
  __syncthreads();
  for (int o = 32; o > 0; o >>= 1) {
    if (tid < o) red[tid] += red[tid + o];
    __syncthreads();
  }
  if (tid == 0) {
    float d = 0.f;
    for (int n = 0; n < SDIM; ++n) d += vfinal[n] * finals[n];
    out[0] = start_prob[0] * __expf(red[0]) * d;
  }
}

// ---------------------------------------------------------------------------
extern "C" void kernel_launch(void* const* d_in, const int* in_sizes, int n_in,
                              void* d_out, int out_size, void* d_ws, size_t ws_size,
                              hipStream_t stream) {
  (void)in_sizes; (void)n_in; (void)out_size; (void)ws_size;
  const int*   seq        = (const int*)d_in[0];
  const float* start_prob = (const float*)d_in[1];
  const float* start_vec  = (const float*)d_in[2];
  const float* trans      = (const float*)d_in[3];
  const float* probt      = (const float*)d_in[4];
  const float* finals     = (const float*)d_in[5];
  float* out = (float*)d_out;

  char* ws = (char*)d_ws;
  _Float16* transA = (_Float16*)(ws + OFF_TA);
  float* prods  = (float*)(ws + OFF_PRODS);
  float* supers = (float*)(ws + OFF_SUPERS);
  float* usuper = (float*)(ws + OFF_USUP);
  float* uvec   = (float*)(ws + OFF_UVEC);
  float* logs   = (float*)(ws + OFF_LOGS);
  float* vfin   = (float*)(ws + OFF_VFIN);

  k_convertA   <<<(128 * MAT_ELEMS) / 256, 256, 0, stream>>>(trans, transA);
  k_chunk_prod <<<CHUNKS, 128, 0, stream>>>(trans, transA, seq, prods);
  k_super_prod <<<NSUP,   128, 0, stream>>>(prods, supers);
  k_super_scan <<<1,       64, 0, stream>>>(start_vec, supers, usuper);
  k_uvec_fill  <<<NSUP,    64, 0, stream>>>(usuper, prods, uvec);
  k_emit       <<<CHUNKS,  64, 0, stream>>>(seq, trans, probt, uvec, logs, vfin);
  k_final      <<<1,       64, 0, stream>>>(logs, vfin, finals, start_prob, out);
}
